// TextRWKV_38302518346042
// MI455X (gfx1250) — matrix operand order repaired and checked by reading, compiled, not run on hardware
//
#include <hip/hip_runtime.h>
#include <hip/hip_bf16.h>

typedef __attribute__((ext_vector_type(16))) __bf16 v16bf;
typedef __attribute__((ext_vector_type(8)))  __bf16 v8bf;
typedef __attribute__((ext_vector_type(8)))  float  v8f;
typedef __attribute__((ext_vector_type(4)))  float  f32x4;

#define Bc   16
#define Tc   1024
#define Dc   1024
#define Hc   16
#define Nch  64
#define BTc  16384
#define EPSc 64e-5f

// ---------------------------------------------------------------------------
// weights: f32 [K,N] -> bf16 transposed [N,K]  (one-time)
// ---------------------------------------------------------------------------
__global__ void convert_transpose_kernel(const float* __restrict__ src,
                                         __bf16* __restrict__ dst,
                                         int K, int N) {
    int i = blockIdx.x * blockDim.x + threadIdx.x;
    if (i < K * N) {
        int n = i / K;
        int k = i - n * K;
        dst[i] = (__bf16)src[(size_t)k * N + n];
    }
}

__device__ __forceinline__ v8bf cvt8(f32x4 a, f32x4 b) {
    v8bf r;
    #pragma unroll
    for (int i = 0; i < 4; ++i) { r[i] = (__bf16)a[i]; r[4 + i] = (__bf16)b[i]; }
    return r;
}

__device__ __forceinline__ float sigmoidf_(float x) {
    return 1.f / (1.f + expf(-x));
}

// ---------------------------------------------------------------------------
// bf16 WMMA GEMM:  C[M,N] (f32) = act(A'[M,K]) @ W[K,N]  (+ epilogue)
//   MIX=1 : A' = x + (x_prev - x) * mix  (token shift, t = row & (Tc-1))
//   PRE   : 0 none, 1 tanh, 2 sigmoid   (applied to A when MIX=0)
//   EP    : 0 store; 1 decay = exp(-exp(-softplus(-(bias+c)) - 0.5));
//           2 sigmoid(bias + c)
// Weights are pre-transposed bf16 [N,K].
// Block: 256 threads (8 waves). Block tile 128x64, wave tile 32x32 (4 WMMA).
// ---------------------------------------------------------------------------
template<int EP>
__device__ __forceinline__ float gemm_epilogue(float c, const float* bias, int col) {
    if (EP == 1) {
        float y = bias[col] + c;
        float w = -log1pf(expf(-y)) - 0.5f;     // -softplus(-y) - 0.5
        return expf(-expf(w));                  // decay
    } else if (EP == 2) {
        return sigmoidf_(bias[col] + c);
    }
    return c;
}

template<int MIX, int PRE, int EP>
__global__ __launch_bounds__(256)
void gemm_wmma_bf16(const float* __restrict__ A,
                    const float* __restrict__ mix,
                    const __bf16* __restrict__ WbT,   // [N,K] bf16
                    float* __restrict__ C,
                    const float* __restrict__ bias,
                    int M, int Nn, int K)
{
    __shared__ __attribute__((aligned(16))) __bf16 lA[128][40];  // 128x32 used
    __shared__ __attribute__((aligned(16))) __bf16 lBt[64][40];  // 64 cols x 32 k

    const int tid    = threadIdx.x;
    const int lane   = tid & 31;
    const int wave   = tid >> 5;
    const int wm     = wave >> 1;        // 0..3 : rows of 32
    const int wn     = wave & 1;         // 0..1 : cols of 32
    const int half   = lane >> 4;
    const int l16    = lane & 15;
    const int blockM = blockIdx.x * 128;
    const int blockN = blockIdx.y * 64;

    v8f acc00 = {}, acc01 = {}, acc10 = {}, acc11 = {};

    for (int k0 = 0; k0 < K; k0 += 32) {
        // ---- stage A tile (128 x 32): 8 consecutive k per thread ----
        #pragma unroll
        for (int seg = 0; seg < 2; ++seg) {
            int chunk = seg * 256 + tid;          // 512 chunks of 8
            int ml   = chunk >> 2;
            int kseg = (chunk & 3) * 8;
            int m    = blockM + ml;
            int kg   = k0 + kseg;
            const float* ap = A + (size_t)m * K + kg;
            f32x4 c0 = *(const f32x4*)(ap);
            f32x4 c1 = *(const f32x4*)(ap + 4);
            if (MIX) {
                int t = m & (Tc - 1);
                f32x4 p0 = {}, p1 = {};
                if (t > 0) {
                    p0 = *(const f32x4*)(ap - K);
                    p1 = *(const f32x4*)(ap - K + 4);
                }
                f32x4 mx0 = *(const f32x4*)(mix + kg);
                f32x4 mx1 = *(const f32x4*)(mix + kg + 4);
                c0 = c0 + (p0 - c0) * mx0;
                c1 = c1 + (p1 - c1) * mx1;
            } else if (PRE == 1) {
                #pragma unroll
                for (int i = 0; i < 4; ++i) { c0[i] = tanhf(c0[i]); c1[i] = tanhf(c1[i]); }
            } else if (PRE == 2) {
                #pragma unroll
                for (int i = 0; i < 4; ++i) { c0[i] = sigmoidf_(c0[i]); c1[i] = sigmoidf_(c1[i]); }
            }
            *(v8bf*)&lA[ml][kseg] = cvt8(c0, c1);
        }
        // ---- stage B tile (64 cols x 32 k) from transposed weights ----
        {
            int nl   = tid >> 2;
            int kseg = (tid & 3) * 8;
            int n    = blockN + nl;
            v8bf vb = {};
            if (n < Nn)
                vb = *(const v8bf*)(WbT + (size_t)n * K + k0 + kseg);
            *(v8bf*)&lBt[nl][kseg] = vb;
        }
        __syncthreads();

        // ---- fragments: two contiguous 16B runs per lane -> ds_load_b128 ----
        const int ar0 = wm * 32 + l16;
        const int ar1 = ar0 + 16;
        v8bf a0lo = *(const v8bf*)&lA[ar0][half * 8];
        v8bf a0hi = *(const v8bf*)&lA[ar0][16 + half * 8];
        v8bf a1lo = *(const v8bf*)&lA[ar1][half * 8];
        v8bf a1hi = *(const v8bf*)&lA[ar1][16 + half * 8];
        v16bf af0 = __builtin_shufflevector(a0lo, a0hi, 0,1,2,3,4,5,6,7,8,9,10,11,12,13,14,15);
        v16bf af1 = __builtin_shufflevector(a1lo, a1hi, 0,1,2,3,4,5,6,7,8,9,10,11,12,13,14,15);

        const int bc0 = wn * 32 + l16;
        const int bc1 = bc0 + 16;
        v8bf b0lo = *(const v8bf*)&lBt[bc0][half * 8];
        v8bf b0hi = *(const v8bf*)&lBt[bc0][16 + half * 8];
        v8bf b1lo = *(const v8bf*)&lBt[bc1][half * 8];
        v8bf b1hi = *(const v8bf*)&lBt[bc1][16 + half * 8];
        v16bf bf0 = __builtin_shufflevector(b0lo, b0hi, 0,1,2,3,4,5,6,7,8,9,10,11,12,13,14,15);
        v16bf bf1 = __builtin_shufflevector(b1lo, b1hi, 0,1,2,3,4,5,6,7,8,9,10,11,12,13,14,15);

        acc00 = __builtin_amdgcn_wmma_f32_16x16x32_bf16(false, af0, false, bf0, (short)0, acc00, false, false);
        acc01 = __builtin_amdgcn_wmma_f32_16x16x32_bf16(false, af0, false, bf1, (short)0, acc01, false, false);
        acc10 = __builtin_amdgcn_wmma_f32_16x16x32_bf16(false, af1, false, bf0, (short)0, acc10, false, false);
        acc11 = __builtin_amdgcn_wmma_f32_16x16x32_bf16(false, af1, false, bf1, (short)0, acc11, false, false);
        __syncthreads();
    }

    // ---- store with epilogue; C layout: VGPR e -> M = e + half*8 ----
    const int col0 = blockN + wn * 32 + l16;
    const int col1 = col0 + 16;
    #pragma unroll
    for (int e = 0; e < 8; ++e) {
        int r0 = blockM + wm * 32 + e + half * 8;
        int r1 = r0 + 16;
        if (col0 < Nn) {
            C[(size_t)r0 * Nn + col0] = gemm_epilogue<EP>(acc00[e], bias, col0);
            C[(size_t)r1 * Nn + col0] = gemm_epilogue<EP>(acc10[e], bias, col0);
        }
        if (col1 < Nn) {
            C[(size_t)r0 * Nn + col1] = gemm_epilogue<EP>(acc01[e], bias, col1);
            C[(size_t)r1 * Nn + col1] = gemm_epilogue<EP>(acc11[e], bias, col1);
        }
    }
}

// ---------------------------------------------------------------------------
// block reduction over 64 threads
// ---------------------------------------------------------------------------
__device__ __forceinline__ float block_reduce64(float x, float* buf, int n) {
    buf[n] = x;
    __syncthreads();
    #pragma unroll
    for (int s = 32; s > 0; s >>= 1) {
        if (n < s) buf[n] += buf[n + s];
        __syncthreads();
    }
    float r = buf[0];
    __syncthreads();
    return r;
}

// ---------------------------------------------------------------------------
// kk = normalize(k * k_k) per head;  k <- k * (1 + (a-1)*k_a)   (in place)
// ---------------------------------------------------------------------------
__global__ __launch_bounds__(64)
void kk_kernel(float* __restrict__ k, const float* __restrict__ a,
               const float* __restrict__ k_k, const float* __restrict__ k_a,
               float* __restrict__ kkout)
{
    __shared__ float buf[64];
    int blk = blockIdx.x;            // (b*T+t)*H + h
    int h   = blk & (Hc - 1);
    int n   = threadIdx.x;
    size_t idx = (size_t)blk * Nch + n;
    int d = h * Nch + n;

    float kv  = k[idx];
    float kkv = kv * k_k[d];
    float ss  = block_reduce64(kkv * kkv, buf, n);
    float nrm = sqrtf(ss);
    kkout[idx] = kkv / fmaxf(nrm, 1e-12f);

    float av = a[idx];
    k[idx] = kv * (1.f + (av - 1.f) * k_a[d]);
}

// ---------------------------------------------------------------------------
// RWKV7 scan. grid = B*H blocks, 64 threads. Thread i owns row i of S (VGPRs).
// ---------------------------------------------------------------------------
__global__ __launch_bounds__(64)
void scan_kernel(const float* __restrict__ r,  const float* __restrict__ dec,
                 const float* __restrict__ k,  const float* __restrict__ v,
                 const float* __restrict__ kk, const float* __restrict__ a,
                 float* __restrict__ o)
{
    int bh = blockIdx.x;
    int b  = bh / Hc;
    int h  = bh % Hc;
    int i  = threadIdx.x;

    float S[Nch];
    #pragma unroll
    for (int j = 0; j < Nch; ++j) S[j] = 0.f;

    __shared__ float sr[Nch], sd[Nch], sk[Nch], sv[Nch], skk[Nch], sa[Nch];

    for (int t = 0; t < Tc; ++t) {
        size_t idx = (((size_t)b * Tc + t) * Hc + h) * Nch + i;
        sr[i]  = r[idx];  sd[i] = dec[idx]; sk[i] = k[idx];
        sv[i]  = v[idx];  skk[i] = kk[idx]; sa[i] = a[idx];
        __syncthreads();

        float sacc = 0.f;
        #pragma unroll
        for (int j = 0; j < Nch; ++j) sacc += S[j] * skk[j];
        sacc = -sacc;                       // S @ a_t, a_t = -kk

        float vi   = sv[i];
        float oacc = 0.f;
        #pragma unroll
        for (int j = 0; j < Nch; ++j) {
            float bj = skk[j] * sa[j];      // b_t = kk * a
            S[j] = S[j] * sd[j] + sacc * bj + vi * sk[j];
            oacc += S[j] * sr[j];
        }
        o[idx] = oacc;
        __syncthreads();
    }
}

// ---------------------------------------------------------------------------
// GroupNorm (per head) + bonus + gate, in place on o. grid = BT*H, 64 thr
// ---------------------------------------------------------------------------
__global__ __launch_bounds__(64)
void post_kernel(float* __restrict__ o, const float* __restrict__ r,
                 const float* __restrict__ k, const float* __restrict__ v,
                 const float* __restrict__ g, const float* __restrict__ rk,
                 const float* __restrict__ gnw, const float* __restrict__ gnb)
{
    __shared__ float buf[64];
    int blk = blockIdx.x;               // (b*T+t)*H + h
    int h   = blk & (Hc - 1);
    int row = blk / Hc;                 // b*T + t
    int n   = threadIdx.x;
    size_t idx = (size_t)blk * Nch + n;
    int d = h * Nch + n;

    float ov   = o[idx];
    float mean = block_reduce64(ov, buf, n) * (1.f / Nch);
    float msq  = block_reduce64(ov * ov, buf, n) * (1.f / Nch);
    float var  = msq - mean * mean;
    float xn   = (ov - mean) * rsqrtf(var + EPSc);
    xn = xn * gnw[d] + gnb[d];

    float rv = r[idx], kv = k[idx];
    float bonus = block_reduce64(rv * kv * rk[d], buf, n);
    xn += bonus * v[idx];

    xn *= g[(size_t)row * Dc + d];
    o[idx] = xn;
}

// ---------------------------------------------------------------------------
// head: out[b,:] = relu(max_t y[b,t,:]) @ W_cls + b_cls.  grid = B, 256 thr
// ---------------------------------------------------------------------------
__global__ __launch_bounds__(256)
void head_kernel(const float* __restrict__ y, const float* __restrict__ Wcls,
                 const float* __restrict__ bcls, float* __restrict__ out)
{
    __shared__ float s0[256], s1[256];
    int b   = blockIdx.x;
    int tid = threadIdx.x;
    float acc0 = 0.f, acc1 = 0.f;
    for (int c = tid; c < Dc; c += 256) {
        const float* p = y + (size_t)b * Tc * Dc + c;
        float mx = -3.4e38f;
        for (int t = 0; t < Tc; ++t) mx = fmaxf(mx, p[(size_t)t * Dc]);
        mx = fmaxf(mx, 0.f);
        acc0 += mx * Wcls[c * 2 + 0];
        acc1 += mx * Wcls[c * 2 + 1];
    }
    s0[tid] = acc0; s1[tid] = acc1;
    __syncthreads();
    for (int s = 128; s > 0; s >>= 1) {
        if (tid < s) { s0[tid] += s0[tid + s]; s1[tid] += s1[tid + s]; }
        __syncthreads();
    }
    if (tid == 0) {
        out[b * 2 + 0] = s0[0] + bcls[0];
        out[b * 2 + 1] = s1[0] + bcls[1];
    }
}

// ---------------------------------------------------------------------------
extern "C" void kernel_launch(void* const* d_in, const int* in_sizes, int n_in,
                              void* d_out, int out_size, void* d_ws, size_t ws_size,
                              hipStream_t stream)
{
    const float* x    = (const float*)d_in[0];
    const float* x_r  = (const float*)d_in[1];
    const float* x_w  = (const float*)d_in[2];
    const float* x_k  = (const float*)d_in[3];
    const float* x_v  = (const float*)d_in[4];
    const float* x_a  = (const float*)d_in[5];
    const float* x_g  = (const float*)d_in[6];
    const float* W_r  = (const float*)d_in[7];
    const float* W_k  = (const float*)d_in[8];
    const float* W_v  = (const float*)d_in[9];
    const float* W_o  = (const float*)d_in[10];
    const float* w0   = (const float*)d_in[11];
    const float* w1   = (const float*)d_in[12];
    const float* w2   = (const float*)d_in[13];
    const float* a0   = (const float*)d_in[14];
    const float* a1   = (const float*)d_in[15];
    const float* a2   = (const float*)d_in[16];
    const float* g1   = (const float*)d_in[17];
    const float* g2   = (const float*)d_in[18];
    const float* k_k  = (const float*)d_in[19];
    const float* k_a  = (const float*)d_in[20];
    const float* r_k  = (const float*)d_in[21];
    const float* gn_w = (const float*)d_in[22];
    const float* gn_b = (const float*)d_in[23];
    const float* Wcls = (const float*)d_in[24];
    const float* bcls = (const float*)d_in[25];

    char*  ws  = (char*)d_ws;
    size_t off = 0;
    auto alloc = [&](size_t bytes) -> void* {
        void* p = ws + off;
        off = (off + bytes + 255) & ~(size_t)255;
        return p;
    };

    __bf16* Wr_b = (__bf16*)alloc((size_t)Dc * Dc * 2);
    __bf16* Wk_b = (__bf16*)alloc((size_t)Dc * Dc * 2);
    __bf16* Wv_b = (__bf16*)alloc((size_t)Dc * Dc * 2);
    __bf16* Wo_b = (__bf16*)alloc((size_t)Dc * Dc * 2);
    __bf16* w1_b = (__bf16*)alloc((size_t)Dc * 64 * 2);
    __bf16* w2_b = (__bf16*)alloc((size_t)64 * Dc * 2);
    __bf16* a1_b = (__bf16*)alloc((size_t)Dc * 64 * 2);
    __bf16* a2_b = (__bf16*)alloc((size_t)64 * Dc * 2);
    __bf16* g1_b = (__bf16*)alloc((size_t)Dc * 160 * 2);
    __bf16* g2_b = (__bf16*)alloc((size_t)160 * Dc * 2);

    float* r_buf    = (float*)alloc((size_t)BTc * Dc * 4);
    float* k_buf    = (float*)alloc((size_t)BTc * Dc * 4);
    float* v_buf    = (float*)alloc((size_t)BTc * Dc * 4);
    float* dec_buf  = (float*)alloc((size_t)BTc * Dc * 4);
    float* a_buf    = (float*)alloc((size_t)BTc * Dc * 4);
    float* g_buf    = (float*)alloc((size_t)BTc * Dc * 4);
    float* kk_buf   = (float*)alloc((size_t)BTc * Dc * 4);
    float* o_buf    = (float*)alloc((size_t)BTc * Dc * 4);
    float* lora_buf = (float*)alloc((size_t)BTc * 160 * 4);
    float* y_buf    = r_buf;   // r is dead after post_kernel

    // one-time transposed bf16 weight conversion:  dst[n*K+k] = src[k*N+n]
    auto convT = [&](const float* s, __bf16* d, int K, int N) {
        int n = K * N;
        convert_transpose_kernel<<<(n + 255) / 256, 256, 0, stream>>>(s, d, K, N);
    };
    convT(W_r, Wr_b, Dc, Dc);   convT(W_k, Wk_b, Dc, Dc);
    convT(W_v, Wv_b, Dc, Dc);   convT(W_o, Wo_b, Dc, Dc);
    convT(w1, w1_b, Dc, 64);    convT(w2, w2_b, 64, Dc);
    convT(a1, a1_b, Dc, 64);    convT(a2, a2_b, 64, Dc);
    convT(g1, g1_b, Dc, 160);   convT(g2, g2_b, 160, Dc);

    dim3 blk(256);
    auto grid_of = [&](int M, int Nn) { return dim3(M / 128, (Nn + 63) / 64); };

    // projections with fused token shift  (MIX=1, PRE=0, EP=0)
    gemm_wmma_bf16<1,0,0><<<grid_of(BTc, Dc), blk, 0, stream>>>(x, x_r, Wr_b, r_buf, nullptr, BTc, Dc, Dc);
    gemm_wmma_bf16<1,0,0><<<grid_of(BTc, Dc), blk, 0, stream>>>(x, x_k, Wk_b, k_buf, nullptr, BTc, Dc, Dc);
    gemm_wmma_bf16<1,0,0><<<grid_of(BTc, Dc), blk, 0, stream>>>(x, x_v, Wv_b, v_buf, nullptr, BTc, Dc, Dc);

    // decay LoRA: stage1 (token shift), stage2 tanh-preact + decay epilogue
    gemm_wmma_bf16<1,0,0><<<grid_of(BTc, 64), blk, 0, stream>>>(x, x_w, w1_b, lora_buf, nullptr, BTc, 64, Dc);
    gemm_wmma_bf16<0,1,1><<<grid_of(BTc, Dc), blk, 0, stream>>>(lora_buf, nullptr, w2_b, dec_buf, w0, BTc, Dc, 64);

    // a LoRA: stage1 (token shift), stage2 sigmoid(bias+c) epilogue
    gemm_wmma_bf16<1,0,0><<<grid_of(BTc, 64), blk, 0, stream>>>(x, x_a, a1_b, lora_buf, nullptr, BTc, 64, Dc);
    gemm_wmma_bf16<0,0,2><<<grid_of(BTc, Dc), blk, 0, stream>>>(lora_buf, nullptr, a2_b, a_buf, a0, BTc, Dc, 64);

    // gate LoRA: stage1 (token shift), stage2 sigmoid-preact
    gemm_wmma_bf16<1,0,0><<<grid_of(BTc, 160), blk, 0, stream>>>(x, x_g, g1_b, lora_buf, nullptr, BTc, 160, Dc);
    gemm_wmma_bf16<0,2,0><<<grid_of(BTc, Dc), blk, 0, stream>>>(lora_buf, nullptr, g2_b, g_buf, nullptr, BTc, Dc, 160);

    kk_kernel<<<BTc * Hc, 64, 0, stream>>>(k_buf, a_buf, k_k, k_a, kk_buf);

    scan_kernel<<<Bc * Hc, 64, 0, stream>>>(r_buf, dec_buf, k_buf, v_buf,
                                            kk_buf, a_buf, o_buf);

    post_kernel<<<BTc * Hc, 64, 0, stream>>>(o_buf, r_buf, k_buf, v_buf,
                                             g_buf, r_k, gn_w, gn_b);

    // output projection (plain)
    gemm_wmma_bf16<0,0,0><<<grid_of(BTc, Dc), blk, 0, stream>>>(o_buf, nullptr, Wo_b, y_buf, nullptr, BTc, Dc, Dc);

    head_kernel<<<Bc, 256, 0, stream>>>(y_buf, Wcls, bcls, (float*)d_out);

    (void)in_sizes; (void)n_in; (void)out_size; (void)ws_size;
}
